// AttentiveFP_70927089926490
// MI455X (gfx1250) — matrix operand order repaired and checked by reading, not compiled, not run on hardware
//
#include <hip/hip_runtime.h>
#include <hip/hip_bf16.h>

// ---------------------------------------------------------------------------
// AttentiveFP forward on gfx1250: all GEMMs via v_wmma_f32_16x16x32_bf16,
// B tiles staged through LDS, A-fragments in contiguous k order (k = 16h + s), matching packed B.
// ---------------------------------------------------------------------------

typedef __attribute__((ext_vector_type(16))) __bf16 bf16x16;
typedef __attribute__((ext_vector_type(8)))  __bf16 bf16x8;
typedef __attribute__((ext_vector_type(8)))  float  f32x8;

__device__ __forceinline__ __bf16 f2bf(float f) {
    unsigned u = __float_as_uint(f);
    unsigned r = (u + 0x7FFFu + ((u >> 16) & 1u)) >> 16;
    unsigned short h = (unsigned short)r;
    __bf16 b;
    __builtin_memcpy(&b, &h, 2);
    return b;
}
__device__ __forceinline__ float bf2f(__bf16 b) {
    unsigned short h;
    __builtin_memcpy(&h, &b, 2);
    return __uint_as_float(((unsigned)h) << 16);
}

// ---------------------------------------------------------------------------
// Pack f32 weight (K x N, optionally transposed source (N x K)) into the
// per-lane B-fragment order this file uses for V_WMMA_*_16x16x32_bf16 (contiguous k):
//   tile(kt,nt): lane l holds B[kt*32 + (l>>4)*16 + e, nt*16 + (l&15)], e=0..15
// out element index = ((kt*(N/16)+nt)*32 + lane)*16 + e   (== K*N total)
// ---------------------------------------------------------------------------
__global__ void __launch_bounds__(256)
pack_weight(const float* __restrict__ W, __bf16* __restrict__ out,
            int K, int N, int transposed) {
    int idx = blockIdx.x * 256 + threadIdx.x;
    if (idx >= K * N) return;
    int e    = idx & 15;
    int lane = (idx >> 4) & 31;
    int tile = idx >> 9;
    int ntn  = N >> 4;
    int kt = tile / ntn, nt = tile % ntn;
    int k = kt * 32 + (lane >> 4) * 16 + e;
    int n = nt * 16 + (lane & 15);
    float w = transposed ? W[(size_t)n * K + k] : W[(size_t)k * N + n];
    out[idx] = f2bf(w);
}

// ---------------------------------------------------------------------------
// WMMA GEMM: C[M x N] = act( A[M x K] @ B[K x N] + bias ), bf16 in, f32 acc.
// A split in K: k <  Ksplit -> A0 row (gidx ? gidx[m] : m), row stride Ksplit
//               k >= Ksplit -> A1 row m, row stride K-Ksplit
// B pre-packed (see pack_weight), staged via LDS (8 KB / block).
// Block: 256 thr = 8 waves; wave w -> rows [blk*128+w*16, +16); cols 128/blk.
// act: 0 = none, 1 = leaky_relu(0.01)
// ---------------------------------------------------------------------------
__global__ void __launch_bounds__(256)
gemm_wmma_bf16(const __bf16* __restrict__ A0, const int* __restrict__ gidx,
               const __bf16* __restrict__ A1, int Ksplit, int K,
               const __bf16* __restrict__ Bp, const float* __restrict__ bias,
               float* __restrict__ Cf, __bf16* __restrict__ Cb,
               int M, int N, int act) {
    __shared__ __align__(32) __bf16 Bsh[8 * 512];   // 8 tiles of 32x16 bf16

    const int lane    = threadIdx.x & 31;
    const int wave    = threadIdx.x >> 5;
    const int halfSel = lane >> 4;          // 0|1: which K/M half this lane covers
    const int lr      = lane & 15;
    const int m0      = blockIdx.x * 128 + wave * 16;
    const int n0      = blockIdx.y * 128;
    const int ntilesN = N >> 4;

    int mrow = m0 + lr;
    if (mrow >= M) mrow = M - 1;            // clamp loads; stores are guarded
    const int rowA0 = gidx ? gidx[mrow] : mrow;
    const int KA1   = K - Ksplit;

    f32x8 acc[8];
#pragma unroll
    for (int i = 0; i < 8; ++i)
#pragma unroll
        for (int j = 0; j < 8; ++j) acc[i][j] = 0.0f;

    for (int kb = 0; kb < K; kb += 32) {
        // ---- stage 8 B tiles (32x128 slab) into LDS, fragment order ----
        __syncthreads();
        {
            const __bf16* bg = Bp +
                ((size_t)(kb >> 5) * ntilesN + (size_t)blockIdx.y * 8) * 512;
            *(bf16x16*)(Bsh + threadIdx.x * 16) =
                *(const bf16x16*)(bg + (size_t)threadIdx.x * 16);
        }
        __syncthreads();

        // ---- A fragment: lane covers K = kb+hs*16+[0..15]
        int kLo = kb + halfSel * 16;
        int kHi = kLo + 8;
        const __bf16* pLo = (kLo < Ksplit)
            ? (A0 + (size_t)rowA0 * Ksplit + kLo)
            : (A1 + (size_t)mrow * KA1 + (kLo - Ksplit));
        const __bf16* pHi = (kHi < Ksplit)
            ? (A0 + (size_t)rowA0 * Ksplit + kHi)
            : (A1 + (size_t)mrow * KA1 + (kHi - Ksplit));
        bf16x8 alo = *(const bf16x8*)pLo;
        bf16x8 ahi = *(const bf16x8*)pHi;
        bf16x16 a;
#pragma unroll
        for (int i = 0; i < 8; ++i) { a[i] = alo[i]; a[8 + i] = ahi[i]; }

        // prefetch next K-slab of A (global_prefetch_b8)
        if (kb + 32 < K) {
            int kp = kb + 32 + halfSel * 8;
            const __bf16* pp = (kp < Ksplit)
                ? (A0 + (size_t)rowA0 * Ksplit + kp)
                : (A1 + (size_t)mrow * KA1 + (kp - Ksplit));
            __builtin_prefetch(pp, 0, 0);
        }

#pragma unroll
        for (int nt = 0; nt < 8; ++nt) {
            bf16x16 b = *(const bf16x16*)(Bsh + nt * 512 + lane * 16);
            acc[nt] = __builtin_amdgcn_wmma_f32_16x16x32_bf16(
                false, a, false, b, (short)0, acc[nt], false, false);
        }
    }

    // ---- epilogue: bias + activation + guarded stores (C layout §7.12.2) ----
#pragma unroll
    for (int nt = 0; nt < 8; ++nt) {
        int n = n0 + nt * 16 + lr;
        float bv = bias ? bias[n] : 0.0f;
#pragma unroll
        for (int r = 0; r < 8; ++r) {
            int m = m0 + halfSel * 8 + r;
            if (m < M) {
                float v = acc[nt][r] + bv;
                if (act == 1) v = v > 0.0f ? v : 0.01f * v;
                size_t off = (size_t)m * N + n;
                if (Cf) Cf[off] = v;
                if (Cb) Cb[off] = f2bf(v);
            }
        }
    }
}

// ---------------------------------------------------------------------------
// score[m] = lrelu( dot([X0[m], X1[idx1?idx1[m]:m]], Wa) + ba ), wave per row.
// ---------------------------------------------------------------------------
__global__ void __launch_bounds__(256)
attn_score(const __bf16* __restrict__ X0, const __bf16* __restrict__ X1,
           const int* __restrict__ idx1, const float* __restrict__ Wa,
           const float* __restrict__ ba, float* __restrict__ out, int M) {
    int wid  = (blockIdx.x * 256 + threadIdx.x) >> 5;
    int lane = threadIdx.x & 31;
    if (wid >= M) return;
    int r1 = idx1 ? idx1[wid] : wid;
    const __bf16* x0 = X0 + (size_t)wid * 128 + lane * 4;
    const __bf16* x1 = X1 + (size_t)r1  * 128 + lane * 4;
    float sum = 0.0f;
#pragma unroll
    for (int i = 0; i < 4; ++i) sum += bf2f(x0[i]) * Wa[lane * 4 + i];
#pragma unroll
    for (int i = 0; i < 4; ++i) sum += bf2f(x1[i]) * Wa[128 + lane * 4 + i];
#pragma unroll
    for (int off = 16; off; off >>= 1) sum += __shfl_xor(sum, off, 32);
    if (lane == 0) {
        float v = sum + ba[0];
        out[wid] = v > 0.0f ? v : 0.01f * v;
    }
}

__global__ void __launch_bounds__(256)
fill_f32(float* __restrict__ p, float v, long n) {
    long i = (long)blockIdx.x * 256 + threadIdx.x;
    if (i < n) p[i] = v;
}

// float atomic-max via signed/unsigned int trick (works with -inf init)
__global__ void __launch_bounds__(256)
seg_atomic_max(const float* __restrict__ x, const int* __restrict__ seg,
               float* __restrict__ m, int M) {
    int i = blockIdx.x * 256 + threadIdx.x;
    if (i >= M) return;
    float v = x[i];
    int s = seg[i];
    if (v >= 0.0f) atomicMax((int*)&m[s], __float_as_int(v));
    else           atomicMin((unsigned int*)&m[s], __float_as_uint(v));
}

__global__ void __launch_bounds__(256)
seg_expsum(float* __restrict__ x, const int* __restrict__ seg,
           const float* __restrict__ m, float* __restrict__ s, int M) {
    int i = blockIdx.x * 256 + threadIdx.x;
    if (i >= M) return;
    float e = __expf(x[i] - m[seg[i]]);
    x[i] = e;
    atomicAdd(&s[seg[i]], e);
}

__global__ void __launch_bounds__(256)
seg_norm(const float* __restrict__ e, const int* __restrict__ seg,
         const float* __restrict__ s, float* __restrict__ att, int M) {
    int i = blockIdx.x * 256 + threadIdx.x;
    if (i >= M) return;
    att[i] = e[i] / s[seg[i]];
}

// out[seg[e]*128 + h] += (a ? a[e] : 1) * x[e*128 + h]
__global__ void __launch_bounds__(256)
scale_scatter_add(const float* __restrict__ x, const float* __restrict__ a,
                  const int* __restrict__ seg, float* __restrict__ out, long MH) {
    long i = (long)blockIdx.x * 256 + threadIdx.x;
    if (i >= MH) return;
    long e = i >> 7;
    int  h = (int)(i & 127);
    float v = x[i];
    if (a) v *= a[e];
    atomicAdd(&out[(size_t)seg[e] * 128 + h], v);
}

// op: 0 copy, 1 lrelu, 2 elu, 3 relu; optional f32 and bf16 outputs
__global__ void __launch_bounds__(256)
unary_map(const float* __restrict__ in, float* __restrict__ of,
          __bf16* __restrict__ ob, long n, int op) {
    long i = (long)blockIdx.x * 256 + threadIdx.x;
    if (i >= n) return;
    float v = in[i];
    if (op == 1)      v = v > 0.0f ? v : 0.01f * v;
    else if (op == 2) v = v > 0.0f ? v : (__expf(v) - 1.0f);
    else if (op == 3) v = fmaxf(v, 0.0f);
    if (of) of[i] = v;
    if (ob) ob[i] = f2bf(v);
}

// GRU combine + relu; gi/gh: M x 384 (pre-biased), h: M x 128
__global__ void __launch_bounds__(256)
gru_combine(const float* __restrict__ gi, const float* __restrict__ gh,
            const float* __restrict__ h, float* __restrict__ of,
            __bf16* __restrict__ ob, long MH) {
    long i = (long)blockIdx.x * 256 + threadIdx.x;
    if (i >= MH) return;
    long m  = i >> 7;
    int  hh = (int)(i & 127);
    const float* gim = gi + (size_t)m * 384;
    const float* ghm = gh + (size_t)m * 384;
    float r = 1.0f / (1.0f + __expf(-(gim[hh] + ghm[hh])));
    float z = 1.0f / (1.0f + __expf(-(gim[128 + hh] + ghm[128 + hh])));
    float nn = tanhf(gim[256 + hh] + r * ghm[256 + hh]);
    float v = (1.0f - z) * nn + z * h[i];
    v = fmaxf(v, 0.0f);
    if (of) of[i] = v;
    if (ob) ob[i] = f2bf(v);
}

__global__ void __launch_bounds__(256)
finalize_out(const float* __restrict__ sup, const float* __restrict__ aw,
             float* __restrict__ out, int GH, int N) {
    int i = blockIdx.x * 256 + threadIdx.x;
    if (i < GH) out[i] = sup[i];
    else if (i < GH + N) out[i] = aw[i - GH];
}

// ---------------------------------------------------------------------------
extern "C" void kernel_launch(void* const* d_in, const int* in_sizes, int n_in,
                              void* d_out, int out_size, void* d_ws, size_t ws_size,
                              hipStream_t stream) {
    (void)in_sizes; (void)n_in; (void)out_size; (void)ws_size;
    const int N = 30000, E = 300000, G = 1000, H = 128, L = 4, T = 4;
    const int H2 = 2 * H, H3 = 3 * H;

    const float* in_node = (const float*)d_in[0];
    const float* in_edge = (const float*)d_in[1];
    const int*   src     = (const int*)d_in[2];
    const int*   dst     = (const int*)d_in[3];
    const int*   gids    = (const int*)d_in[4];
    const float* aWn  = (const float*)d_in[5];
    const float* abn  = (const float*)d_in[6];
    const float* aWe  = (const float*)d_in[7];
    const float* abe  = (const float*)d_in[8];
    const float* aWa  = (const float*)d_in[9];
    const float* aba  = (const float*)d_in[10];
    const float* aWt  = (const float*)d_in[11];
    const float* abt  = (const float*)d_in[12];
    const float* aWih = (const float*)d_in[13];
    const float* aWhh = (const float*)d_in[14];
    const float* abih = (const float*)d_in[15];
    const float* abhh = (const float*)d_in[16];
    const float* mWa  = (const float*)d_in[17];
    const float* mba  = (const float*)d_in[18];
    const float* mWt  = (const float*)d_in[19];
    const float* mbt  = (const float*)d_in[20];
    const float* mWih = (const float*)d_in[21];
    const float* mWhh = (const float*)d_in[22];
    const float* mbih = (const float*)d_in[23];
    const float* mbhh = (const float*)d_in[24];
    float* out = (float*)d_out;

    // ---------------- workspace carve (bump allocator) ----------------
    char* ws = (char*)d_ws;
    size_t off = 0;
    auto alloc = [&](size_t bytes) -> void* {
        void* p = ws + off;
        off += (bytes + 255) & ~(size_t)255;
        return p;
    };
    __bf16* node_b = (__bf16*)alloc((size_t)N * H * 2);
    float*  node_f = (float*) alloc((size_t)N * H * 4);
    __bf16* ne_b   = (__bf16*)alloc((size_t)N * H * 2);
    float*  ne_f   = (float*) alloc((size_t)N * H * 4);
    __bf16* edge_b = (__bf16*)alloc((size_t)E * H * 2);
    __bf16* msg_b  = (__bf16*)alloc((size_t)E * H * 2);
    float*  big    = (float*) alloc((size_t)E * H * 4);  // hn, aliased gi/gh
    float*  gi     = big;                                // lifetime-disjoint
    float*  gh     = big + (size_t)N * H3;
    float*  score  = (float*) alloc((size_t)E * 4);
    float*  att    = (float*) alloc((size_t)E * 4);
    float*  mbuf   = (float*) alloc((size_t)N * 4);
    float*  sbuf   = (float*) alloc((size_t)N * 4);
    float*  ctx_f  = (float*) alloc((size_t)N * H * 4);
    __bf16* ctx_b  = (__bf16*)alloc((size_t)N * H * 2);
    float*  supr   = (float*) alloc((size_t)G * H * 4);
    float*  s_f    = (float*) alloc((size_t)G * H * 4);
    __bf16* s_b    = (__bf16*)alloc((size_t)G * H * 2);
    __bf16* sctx_b = (__bf16*)alloc((size_t)G * H * 2);
    __bf16* pk_aWn  = (__bf16*)alloc((size_t)L * H * H * 2);
    __bf16* pk_aWe  = (__bf16*)alloc((size_t)L * H2 * H * 2);
    __bf16* pk_aWt  = (__bf16*)alloc((size_t)L * H * H * 2);
    __bf16* pk_aWih = (__bf16*)alloc((size_t)L * H * H3 * 2);
    __bf16* pk_aWhh = (__bf16*)alloc((size_t)L * H * H3 * 2);
    __bf16* pk_mWt  = (__bf16*)alloc((size_t)T * H * H * 2);
    __bf16* pk_mWih = (__bf16*)alloc((size_t)T * H * H3 * 2);
    __bf16* pk_mWhh = (__bf16*)alloc((size_t)T * H * H3 * 2);

    auto ew = [](long n) { return dim3((unsigned)((n + 255) / 256)); };
    const dim3 blk(256);
    auto ggrid = [](int M, int Nn) {
        return dim3((unsigned)((M + 127) / 128), (unsigned)(Nn / 128));
    };

    // ---------------- pack weights into WMMA B-fragment order ----------------
    for (int l = 0; l < L; ++l) {
        pack_weight<<<ew((long)H * H), blk, 0, stream>>>(aWn + (size_t)l * H * H, pk_aWn + (size_t)l * H * H, H, H, 0);
        pack_weight<<<ew((long)H2 * H), blk, 0, stream>>>(aWe + (size_t)l * H2 * H, pk_aWe + (size_t)l * H2 * H, H2, H, 0);
        pack_weight<<<ew((long)H * H), blk, 0, stream>>>(aWt + (size_t)l * H * H, pk_aWt + (size_t)l * H * H, H, H, 0);
        pack_weight<<<ew((long)H * H3), blk, 0, stream>>>(aWih + (size_t)l * H3 * H, pk_aWih + (size_t)l * H * H3, H, H3, 1);
        pack_weight<<<ew((long)H * H3), blk, 0, stream>>>(aWhh + (size_t)l * H3 * H, pk_aWhh + (size_t)l * H * H3, H, H3, 1);
    }
    for (int t = 0; t < T; ++t) {
        pack_weight<<<ew((long)H * H), blk, 0, stream>>>(mWt + (size_t)t * H * H, pk_mWt + (size_t)t * H * H, H, H, 0);
        pack_weight<<<ew((long)H * H3), blk, 0, stream>>>(mWih + (size_t)t * H3 * H, pk_mWih + (size_t)t * H * H3, H, H3, 1);
        pack_weight<<<ew((long)H * H3), blk, 0, stream>>>(mWhh + (size_t)t * H3 * H, pk_mWhh + (size_t)t * H * H3, H, H3, 1);
    }

    // ---------------- cast inputs to bf16 ----------------
    unary_map<<<ew((long)N * H), blk, 0, stream>>>(in_node, nullptr, node_b, (long)N * H, 0);
    unary_map<<<ew((long)E * H), blk, 0, stream>>>(in_edge, nullptr, edge_b, (long)E * H, 0);

    // ---------------- L message-passing layers ----------------
    for (int l = 0; l < L; ++l) {
        // ne = lrelu(node @ aWn + abn)
        gemm_wmma_bf16<<<ggrid(N, H), blk, 0, stream>>>(
            node_b, nullptr, nullptr, H, H, pk_aWn + (size_t)l * H * H,
            abn + (size_t)l * H, ne_f, ne_b, N, H, 1);
        // msg = lrelu([ne[src], edge] @ aWe + abe)   (gather+concat fused)
        gemm_wmma_bf16<<<ggrid(E, H), blk, 0, stream>>>(
            ne_b, src, edge_b, H, H2, pk_aWe + (size_t)l * H2 * H,
            abe + (size_t)l * H, nullptr, msg_b, E, H, 1);
        // score = lrelu([msg, ne[dst]] @ aWa + aba)
        attn_score<<<ew((long)E * 32), blk, 0, stream>>>(
            msg_b, ne_b, dst, aWa + (size_t)l * H2, aba + l, score, E);
        // att = segment_softmax(score, dst)
        fill_f32<<<ew(N), blk, 0, stream>>>(mbuf, -__builtin_inff(), N);
        fill_f32<<<ew(N), blk, 0, stream>>>(sbuf, 0.0f, N);
        seg_atomic_max<<<ew(E), blk, 0, stream>>>(score, dst, mbuf, E);
        seg_expsum<<<ew(E), blk, 0, stream>>>(score, dst, mbuf, sbuf, E);
        seg_norm<<<ew(E), blk, 0, stream>>>(score, dst, sbuf, att, E);
        // hn = msg @ aWt + abt
        gemm_wmma_bf16<<<ggrid(E, H), blk, 0, stream>>>(
            msg_b, nullptr, nullptr, H, H, pk_aWt + (size_t)l * H * H,
            abt + (size_t)l * H, big, nullptr, E, H, 0);
        // ctx = elu(segment_sum(att * hn, dst))
        fill_f32<<<ew((long)N * H), blk, 0, stream>>>(ctx_f, 0.0f, (long)N * H);
        scale_scatter_add<<<ew((long)E * H), blk, 0, stream>>>(big, att, dst, ctx_f, (long)E * H);
        unary_map<<<ew((long)N * H), blk, 0, stream>>>(ctx_f, ctx_f, ctx_b, (long)N * H, 2);
        // GRU: gi = ctx @ Wih^T + bih ; gh = ne @ Whh^T + bhh
        gemm_wmma_bf16<<<ggrid(N, H3), blk, 0, stream>>>(
            ctx_b, nullptr, nullptr, H, H, pk_aWih + (size_t)l * H * H3,
            abih + (size_t)l * H3, gi, nullptr, N, H3, 0);
        gemm_wmma_bf16<<<ggrid(N, H3), blk, 0, stream>>>(
            ne_b, nullptr, nullptr, H, H, pk_aWhh + (size_t)l * H * H3,
            abhh + (size_t)l * H3, gh, nullptr, N, H3, 0);
        // node = relu(gru(ctx, ne))
        gru_combine<<<ew((long)N * H), blk, 0, stream>>>(
            gi, gh, ne_f, node_f, node_b, (long)N * H);
    }

    // ---------------- readout ----------------
    fill_f32<<<ew((long)G * H), blk, 0, stream>>>(supr, 0.0f, (long)G * H);
    scale_scatter_add<<<ew((long)N * H), blk, 0, stream>>>(node_f, nullptr, gids, supr, (long)N * H);

    for (int t = 0; t < T; ++t) {
        // s = lrelu(super_node)
        unary_map<<<ew((long)G * H), blk, 0, stream>>>(supr, s_f, s_b, (long)G * H, 1);
        // sc = lrelu([node, s[gids]] @ mWa + mba)
        attn_score<<<ew((long)N * 32), blk, 0, stream>>>(
            node_b, s_b, gids, mWa + (size_t)t * H2, mba + t, score, N);
        // aw = segment_softmax(sc, gids)
        fill_f32<<<ew(G), blk, 0, stream>>>(mbuf, -__builtin_inff(), G);
        fill_f32<<<ew(G), blk, 0, stream>>>(sbuf, 0.0f, G);
        seg_atomic_max<<<ew(N), blk, 0, stream>>>(score, gids, mbuf, N);
        seg_expsum<<<ew(N), blk, 0, stream>>>(score, gids, mbuf, sbuf, N);
        seg_norm<<<ew(N), blk, 0, stream>>>(score, gids, sbuf, att, N);
        // hn = node @ mWt + mbt
        gemm_wmma_bf16<<<ggrid(N, H), blk, 0, stream>>>(
            node_b, nullptr, nullptr, H, H, pk_mWt + (size_t)t * H * H,
            mbt + (size_t)t * H, big, nullptr, N, H, 0);
        // sctx = elu(segment_sum(aw * hn, gids))   (reuse ctx_f for G rows)
        fill_f32<<<ew((long)G * H), blk, 0, stream>>>(ctx_f, 0.0f, (long)G * H);
        scale_scatter_add<<<ew((long)N * H), blk, 0, stream>>>(big, att, gids, ctx_f, (long)N * H);
        unary_map<<<ew((long)G * H), blk, 0, stream>>>(ctx_f, ctx_f, sctx_b, (long)G * H, 2);
        // GRU on graphs
        gemm_wmma_bf16<<<ggrid(G, H3), blk, 0, stream>>>(
            sctx_b, nullptr, nullptr, H, H, pk_mWih + (size_t)t * H * H3,
            mbih + (size_t)t * H3, gi, nullptr, G, H3, 0);
        gemm_wmma_bf16<<<ggrid(G, H3), blk, 0, stream>>>(
            s_b, nullptr, nullptr, H, H, pk_mWhh + (size_t)t * H * H3,
            mbhh + (size_t)t * H3, gh, nullptr, G, H3, 0);
        gru_combine<<<ew((long)G * H), blk, 0, stream>>>(
            gi, gh, s_f, supr, nullptr, (long)G * H);
    }

    // d_out = [super_node (G*H), att_w (N)]
    finalize_out<<<ew((long)G * H + N), blk, 0, stream>>>(supr, att, out, G * H, N);
}
